// FewTPLinker_64278480552179
// MI455X (gfx1250) — matrix operand order repaired and checked by reading, not compiled, not run on hardware
//
#include <hip/hip_runtime.h>

typedef __bf16 bf16;
typedef __attribute__((ext_vector_type(16))) __bf16 v16bf;
typedef __attribute__((ext_vector_type(8)))  float  v8f;

#define HD 768     // hidden
#define SD 128     // seq len
#define BD 8       // batch
#define MD 128     // head dim per output
#define NP 8256    // S*(S+1)/2 pairs
#define NC 1024    // B*S cond rows
#define N3 384     // 3*M concatenated output cols

// ---------------- WMMA fragment helpers (bf16 16x16x32, wave32) -------------
// A 16x32: lane half=0 -> e0..15 = K0..15 ; half=1 -> K16..31
__device__ __forceinline__ v16bf make_afrag_f32(const float* __restrict__ row,
                                                int k0, int hh) {
  const float* p = row + k0 + hh * 16;
  float4 x0 = *(const float4*)(p);
  float4 x1 = *(const float4*)(p + 4);
  float4 x2 = *(const float4*)(p + 8);
  float4 x3 = *(const float4*)(p + 12);
  v16bf a;
  a[0]=(bf16)x0.x; a[1]=(bf16)x0.y; a[2]=(bf16)x0.z; a[3]=(bf16)x0.w;
  a[4]=(bf16)x1.x; a[5]=(bf16)x1.y; a[6]=(bf16)x1.z; a[7]=(bf16)x1.w;
  a[8]=(bf16)x2.x; a[9]=(bf16)x2.y; a[10]=(bf16)x2.z; a[11]=(bf16)x2.w;
  a[12]=(bf16)x3.x; a[13]=(bf16)x3.y; a[14]=(bf16)x3.z; a[15]=(bf16)x3.w;
  return a;
}

__device__ __forceinline__ v16bf make_afrag_prod(const float* __restrict__ xr,
                                                 const float* __restrict__ gr,
                                                 int k0, int hh) {
  const float* px = xr + k0 + hh * 16;
  const float* pg = gr + k0 + hh * 16;
  float4 x0 = *(const float4*)(px);      float4 g0 = *(const float4*)(pg);
  float4 x1 = *(const float4*)(px + 4);  float4 g1 = *(const float4*)(pg + 4);
  float4 x2 = *(const float4*)(px + 8);  float4 g2 = *(const float4*)(pg + 8);
  float4 x3 = *(const float4*)(px + 12); float4 g3 = *(const float4*)(pg + 12);
  v16bf a;
  a[0]=(bf16)(x0.x*g0.x); a[1]=(bf16)(x0.y*g0.y); a[2]=(bf16)(x0.z*g0.z); a[3]=(bf16)(x0.w*g0.w);
  a[4]=(bf16)(x1.x*g1.x); a[5]=(bf16)(x1.y*g1.y); a[6]=(bf16)(x1.z*g1.z); a[7]=(bf16)(x1.w*g1.w);
  a[8]=(bf16)(x2.x*g2.x); a[9]=(bf16)(x2.y*g2.y); a[10]=(bf16)(x2.z*g2.z); a[11]=(bf16)(x2.w*g2.w);
  a[12]=(bf16)(x3.x*g3.x); a[13]=(bf16)(x3.y*g3.y); a[14]=(bf16)(x3.z*g3.z); a[15]=(bf16)(x3.w*g3.w);
  return a;
}

// ---------------- prep kernels ----------------------------------------------
__global__ void cvt_w_kernel(const float* __restrict__ wg, const float* __restrict__ wb,
                             bf16* __restrict__ wgb, bf16* __restrict__ wbb) {
  int i = blockIdx.x * 256 + threadIdx.x;
  if (i < HD * HD) { wgb[i] = (bf16)wg[i]; wbb[i] = (bf16)wb[i]; }
}

// Bt[n*HD + k] = bf16(Wcat[k, n]) with Wcat = [w_ent | w_head | w_tail] (H x 384)
__global__ void build_bt_kernel(const float* __restrict__ we, const float* __restrict__ wh,
                                const float* __restrict__ wt, const float* __restrict__ be,
                                const float* __restrict__ bh, const float* __restrict__ btl,
                                bf16* __restrict__ Bt, float* __restrict__ bcat) {
  int t = blockIdx.x * 256 + threadIdx.x;
  if (t < N3 * HD) {
    int n = t / HD;
    int k = t - n * HD;
    const float* w = (n < 128) ? we : ((n < 256) ? wh : wt);
    Bt[t] = (bf16)w[(size_t)k * MD + (n & 127)];
  }
  if (t < N3)
    bcat[t] = (t < 128) ? be[t] : ((t < 256) ? bh[t - 128] : btl[t - 256]);
}

// triu pair LUT: p -> (i, j), i<=j, row-major in i then j (jnp.triu_indices order)
__global__ void pairs_kernel(int* __restrict__ iiL, int* __restrict__ jjL) {
  int i = blockIdx.x;
  int off = i * SD - (i * (i - 1)) / 2;
  for (int j = i + (int)threadIdx.x; j < SD; j += blockDim.x) {
    int p = off + (j - i);
    iiL[p] = i;
    jjL[p] = j;
  }
}

// per-row LN stats; note reference uses (var+eps)^2 (not sqrt) as divisor
__global__ void stats_kernel(const float* __restrict__ seq, float* __restrict__ xnorm) {
  int row = blockIdx.x;              // 0..1023 = b*S + s
  int tid = threadIdx.x;
  const float* x = seq + (size_t)row * HD;
  float s = 0.f, ss = 0.f;
  for (int h = tid; h < HD; h += 256) { float v = x[h]; s += v; ss += v * v; }
  __shared__ float sh1[256], sh2[256];
  sh1[tid] = s; sh2[tid] = ss;
  __syncthreads();
  for (int off = 128; off > 0; off >>= 1) {
    if (tid < off) { sh1[tid] += sh1[tid + off]; sh2[tid] += sh2[tid + off]; }
    __syncthreads();
  }
  float mean = sh1[0] * (1.0f / HD);
  float var  = sh2[0] * (1.0f / HD) - mean * mean;
  float d = var + 1e-12f;
  float r = 1.0f / (d * d);
  float* o = xnorm + (size_t)row * HD;
  for (int h = tid; h < HD; h += 256) o[h] = (x[h] - mean) * r;
}

// ---------------- generic wave-tiled WMMA GEMM ------------------------------
// C[r,n] = sum_k bf16(A[r,k]) * Bw[n,k] + bias[n]; K fixed at 768.
// block = 128 threads (4 waves); block tile 64 rows x 64 cols; wave = 16x64.
__global__ void __launch_bounds__(128)
gemm_cond_kernel(const float* __restrict__ A, const bf16* __restrict__ Bw,
                 const float* __restrict__ bias, float* __restrict__ C, int N) {
  const int lane = threadIdx.x & 31;
  const int wave = threadIdx.x >> 5;
  const int hh   = lane >> 4;
  const int l16  = lane & 15;
  const int rowBase = blockIdx.x * 64 + wave * 16;
  const int colBase = blockIdx.y * 64;
  const float* Arow = A + (size_t)(rowBase + l16) * HD;

  v8f z = {0.f,0.f,0.f,0.f,0.f,0.f,0.f,0.f};
  v8f acc[4];
#pragma unroll
  for (int c = 0; c < 4; ++c) acc[c] = z;

  for (int k0 = 0; k0 < HD; k0 += 32) {
    v16bf a = make_afrag_f32(Arow, k0, hh);
#pragma unroll
    for (int c = 0; c < 4; ++c) {
      const bf16* Brow = Bw + (size_t)(colBase + c * 16 + l16) * HD;
      v16bf bfrag = *(const v16bf*)(Brow + k0 + hh * 16);
      acc[c] = __builtin_amdgcn_wmma_f32_16x16x32_bf16(
          false, a, false, bfrag, (short)0, acc[c], false, false);
    }
  }
#pragma unroll
  for (int c = 0; c < 4; ++c) {
    int n = colBase + c * 16 + l16;
    float bn = bias[n];
#pragma unroll
    for (int v = 0; v < 8; ++v) {
      int r = rowBase + hh * 8 + v;
      C[(size_t)r * N + n] = acc[c][v] + bn;
    }
  }
}

// ---------------- fused pair GEMM + tanh + scatter --------------------------
// out[layer,b,p,m] = tanh( (xnorm[b,j(p)]⊙gamma_c[b,i(p)]) @ Wcat[:,n]
//                          + beta_proj[b,i(p),n] ),  n = layer*128+m
__global__ void __launch_bounds__(128)
pair_gemm_kernel(const float* __restrict__ xnorm, const float* __restrict__ gammac,
                 const float* __restrict__ betaproj, const bf16* __restrict__ Bt,
                 const int* __restrict__ iiL, const int* __restrict__ jjL,
                 float* __restrict__ out) {
  __shared__ bf16 ldsB[64 * 32];                 // 64 n-rows x 32 k (4 KB)
  const int lane = threadIdx.x & 31;
  const int wave = threadIdx.x >> 5;
  const int hh   = lane >> 4;
  const int l16  = lane & 15;
  const int rowBase = blockIdx.x * 64 + wave * 16;
  const int colBase = blockIdx.y * 64;
  const int b = blockIdx.z;

  const int pa = rowBase + l16;                  // this lane's A row (pair)
  const float* xr = xnorm  + ((size_t)b * SD + jjL[pa]) * HD;
  const float* gr = gammac + ((size_t)b * SD + iiL[pa]) * HD;

  // B staging assignment: thread -> (n-row, k-half) of the 64x32 tile
  const int tn = threadIdx.x >> 1;
  const int th = threadIdx.x & 1;
  const bf16* bsrc = Bt + (size_t)(colBase + tn) * HD + th * 16;
  bf16* bdst = ldsB + tn * 32 + th * 16;

  v8f z = {0.f,0.f,0.f,0.f,0.f,0.f,0.f,0.f};
  v8f acc[4];
#pragma unroll
  for (int c = 0; c < 4; ++c) acc[c] = z;

  for (int k0 = 0; k0 < HD; k0 += 32) {
    __syncthreads();                             // previous tile fully consumed
    *(v16bf*)bdst = *(const v16bf*)(bsrc + k0);
    __syncthreads();
    v16bf a = make_afrag_prod(xr, gr, k0, hh);
#pragma unroll
    for (int c = 0; c < 4; ++c) {
      v16bf bfrag = *(const v16bf*)(ldsB + (c * 16 + l16) * 32 + hh * 16);
      acc[c] = __builtin_amdgcn_wmma_f32_16x16x32_bf16(
          false, a, false, bfrag, (short)0, acc[c], false, false);
    }
  }

  int prow[8], irow[8];
#pragma unroll
  for (int v = 0; v < 8; ++v) {
    prow[v] = rowBase + hh * 8 + v;
    irow[v] = iiL[prow[v]];
  }
#pragma unroll
  for (int c = 0; c < 4; ++c) {
    int n = colBase + c * 16 + l16;
    int layer = n >> 7;
    int m = n & 127;
    size_t obase = (((size_t)layer * BD + b) * NP) * MD + m;
#pragma unroll
    for (int v = 0; v < 8; ++v) {
      float val = acc[c][v] + betaproj[((size_t)b * SD + irow[v]) * N3 + n];
      out[obase + (size_t)prow[v] * MD] = tanhf(val);
    }
  }
}

// ---------------- host-side orchestration -----------------------------------
extern "C" void kernel_launch(void* const* d_in, const int* in_sizes, int n_in,
                              void* d_out, int out_size, void* d_ws, size_t ws_size,
                              hipStream_t stream) {
  const float* seq     = (const float*)d_in[0];   // (B,S,H)
  const float* gamma   = (const float*)d_in[1];   // (H)
  const float* beta    = (const float*)d_in[2];   // (H)
  const float* w_beta  = (const float*)d_in[3];   // (H,H) row-major (o,h)
  const float* w_gamma = (const float*)d_in[4];   // (H,H)
  const float* w_ent   = (const float*)d_in[5];   // (H,M)
  const float* b_ent   = (const float*)d_in[6];
  const float* w_head  = (const float*)d_in[7];
  const float* b_head  = (const float*)d_in[8];
  const float* w_tail  = (const float*)d_in[9];
  const float* b_tail  = (const float*)d_in[10];
  float* out = (float*)d_out;                     // (3,B,P,M) fp32

  // workspace layout (all offsets 256B aligned)
  char* ws = (char*)d_ws;
  bf16*  wg_bf    = (bf16*)(ws);                          // 768*768   bf16
  bf16*  wb_bf    = (bf16*)(ws + 1179648);                // 768*768   bf16
  bf16*  Bt       = (bf16*)(ws + 2359296);                // 384*768   bf16
  float* gammac   = (float*)(ws + 2949120);               // 1024*768  f32
  float* betac    = (float*)(ws + 6094848);               // 1024*768  f32
  float* betaproj = (float*)(ws + 9240576);               // 1024*384  f32
  float* xnorm    = (float*)(ws + 10813440);              // 1024*768  f32
  int*   iiL      = (int*)(ws + 13959168);                // 8256 int
  int*   jjL      = (int*)(ws + 13992192);                // 8256 int
  float* bcat     = (float*)(ws + 14025216);              // 384 f32

  // 0) weight conversions + LUTs + LN stats
  cvt_w_kernel<<<(HD * HD + 255) / 256, 256, 0, stream>>>(w_gamma, w_beta, wg_bf, wb_bf);
  build_bt_kernel<<<(N3 * HD + 255) / 256, 256, 0, stream>>>(
      w_ent, w_head, w_tail, b_ent, b_head, b_tail, Bt, bcat);
  pairs_kernel<<<SD, 128, 0, stream>>>(iiL, jjL);
  stats_kernel<<<NC, 256, 0, stream>>>(seq, xnorm);

  // 1) cond-row GEMMs: gamma_c / beta_c = seq @ W^T + bias  (1024 x 768, K=768)
  gemm_cond_kernel<<<dim3(NC / 64, HD / 64), 128, 0, stream>>>(seq, wg_bf, gamma, gammac, HD);
  gemm_cond_kernel<<<dim3(NC / 64, HD / 64), 128, 0, stream>>>(seq, wb_bf, beta,  betac,  HD);

  // 2) beta_proj = beta_c @ Wcat + bcat  (1024 x 384, K=768)
  gemm_cond_kernel<<<dim3(NC / 64, N3 / 64), 128, 0, stream>>>(betac, Bt, bcat, betaproj, N3);

  // 3) fused pair GEMM (8256 x 384 per batch, K=768) + tanh + scatter
  pair_gemm_kernel<<<dim3(NP / 64, N3 / 64, BD), 128, 0, stream>>>(
      xnorm, gammac, betaproj, Bt, iiL, jjL, out);
}
